// ClassifierJoint_84361747628356
// MI455X (gfx1250) — matrix operand order repaired and checked by reading, not compiled, not run on hardware
//
#include <hip/hip_runtime.h>
#include <hip/hip_bf16.h>
#include <math.h>

#define N_NODES 16384
#define N_EDGES 262144
#define D_FEAT  1280
#define D_LM    1024
#define D_NF    256

typedef __attribute__((ext_vector_type(16))) __bf16 v16bf;
typedef __attribute__((ext_vector_type(8)))  __bf16 v8bf;
typedef __attribute__((ext_vector_type(4)))  __bf16 v4bf;
typedef __attribute__((ext_vector_type(8)))  float  v8f;

// ---------------------------------------------------------------------------
// 1. concat lm[N,1024] || nf[N,256] -> feat0[N,1280], float4 granularity
// ---------------------------------------------------------------------------
__global__ void concat_kernel(const float* __restrict__ lm,
                              const float* __restrict__ nf,
                              float* __restrict__ out) {
    int i = blockIdx.x * blockDim.x + threadIdx.x;     // over N*320 float4s
    int n  = i / 320;
    int d4 = i - n * 320;
    float4 v;
    if (d4 < 256) v = ((const float4*)lm)[(size_t)n * 256 + d4];
    else          v = ((const float4*)nf)[(size_t)n * 64 + (d4 - 256)];
    ((float4*)out)[i] = v;
}

// ---------------------------------------------------------------------------
// 2. zero a float buffer
// ---------------------------------------------------------------------------
__global__ void zero_kernel(float* __restrict__ p, int n) {
    int i = blockIdx.x * blockDim.x + threadIdx.x;
    if (i < n) p[i] = 0.0f;
}

// ---------------------------------------------------------------------------
// 3. edge scatter: agg[dst] += feat[src] / (ef^2 + 1e-6)
//    one block per edge, 320 threads, float4 gather + 4 hw float atomics
// ---------------------------------------------------------------------------
__global__ void scatter_kernel(const float* __restrict__ feat,
                               const int* __restrict__ src,
                               const int* __restrict__ dst,
                               const float* __restrict__ ef,
                               float* __restrict__ agg) {
    int e = blockIdx.x;
    int s = src[e];
    int t = dst[e];
    float w = ef[e];
    w = 1.0f / (w * w + 1e-6f);
    const float4* fs = (const float4*)(feat + (size_t)s * D_FEAT);
    float* ad = agg + (size_t)t * D_FEAT;
    int i = threadIdx.x;                  // 0..319
    float4 v = fs[i];
    int base = i * 4;
    unsafeAtomicAdd(ad + base + 0, v.x * w);
    unsafeAtomicAdd(ad + base + 1, v.y * w);
    unsafeAtomicAdd(ad + base + 2, v.z * w);
    unsafeAtomicAdd(ad + base + 3, v.w * w);
}

// ---------------------------------------------------------------------------
// 4. x_bf16 = bf16(a + b)   (GIN "(1+eps)*feat + agg" with eps=0)
// ---------------------------------------------------------------------------
__global__ void addcvt_kernel(const float* __restrict__ a,
                              const float* __restrict__ b,
                              __bf16* __restrict__ o) {
    int i = blockIdx.x * blockDim.x + threadIdx.x;    // over N*D/4
    float4 x = ((const float4*)a)[i];
    float4 y = ((const float4*)b)[i];
    v4bf r;
    r.x = (__bf16)(x.x + y.x);
    r.y = (__bf16)(x.y + y.y);
    r.z = (__bf16)(x.z + y.z);
    r.w = (__bf16)(x.w + y.w);
    ((v4bf*)o)[i] = r;
}

// 4b. plain f32 -> bf16 convert (weights)
__global__ void cvt_kernel(const float* __restrict__ a,
                           __bf16* __restrict__ o) {
    int i = blockIdx.x * blockDim.x + threadIdx.x;    // over D*D/4
    float4 x = ((const float4*)a)[i];
    v4bf r;
    r.x = (__bf16)x.x; r.y = (__bf16)x.y; r.z = (__bf16)x.z; r.w = (__bf16)x.w;
    ((v4bf*)o)[i] = r;
}

// ---------------------------------------------------------------------------
// 5. WMMA GEMM: C[N,D] = A[N,D] @ W[D,D]^T + bias, optional +resid,
//    optional store, optional column-sum accumulation (for mean pooling).
//    Block = 256 thr (8 waves). Wave computes a 16(M) x 64(N) strip.
//    grid = (N/128, D/64).
//    B tile (64 rows x 32 K = 4KB, shared by all 8 waves) is staged into
//    LDS with GLOBAL_LOAD_ASYNC_TO_LDS_B128 (ASYNCcnt), double-buffered so
//    the copy of step k+1 overlaps the WMMAs of step k.
// ---------------------------------------------------------------------------
#define BTILE_ELEMS (64 * 32)   // 4096 bf16 = 8KB... no: 64*32*2B = 4KB

__device__ __forceinline__ void stage_b_async(const __bf16* __restrict__ W,
                                              int jBase, int kk,
                                              __bf16* ldsDst) {
    // 256 threads: thread t copies 16B: row = t>>2 (0..63), chunk = t&3 (0..3)
    const int t = threadIdx.x;
    const int row = t >> 2;
    const int chunk = t & 3;
    const __bf16* g = W + (size_t)(jBase + row) * D_FEAT + kk + chunk * 8;
    unsigned ldsOff = (unsigned)(uintptr_t)(ldsDst + row * 32 + chunk * 8);
    asm volatile("global_load_async_to_lds_b128 %0, %1, off"
                 :: "v"(ldsOff), "v"(g) : "memory");
}

__global__ void gemm_bf16_kernel(const __bf16* __restrict__ A,
                                 const __bf16* __restrict__ W,
                                 const float* __restrict__ bias,
                                 float* __restrict__ out,         // nullable
                                 const float* __restrict__ resid, // nullable
                                 float* __restrict__ colsum) {    // nullable
    __shared__ __align__(16) __bf16 bsmem[2][BTILE_ELEMS];

    const int lane = threadIdx.x & 31;
    const int wave = threadIdx.x >> 5;
    const int half = lane >> 4;          // 0 or 1
    const int l16  = lane & 15;

    const int mBase = blockIdx.x * 128 + wave * 16;   // 16 rows of C
    const int jBase = blockIdx.y * 64;                // 64 cols of C

    // A: row m = mBase + l16 ; two 16B chunks per K-step of 32
    const __bf16* arow = A + (size_t)(mBase + l16) * D_FEAT;

    v8f acc[4] = {};

    const int NSTEP = D_FEAT / 32;       // 40
    stage_b_async(W, jBase, 0, &bsmem[0][0]);

    for (int it = 0; it < NSTEP; ++it) {
        const int cur = it & 1;
        const int kk = it * 32;
        const bool more = (it + 1 < NSTEP);
        if (more) {
            stage_b_async(W, jBase, kk + 32, &bsmem[cur ^ 1][0]);
            asm volatile("s_wait_asynccnt 0x1" ::: "memory");
        } else {
            asm volatile("s_wait_asynccnt 0x0" ::: "memory");
        }
        __syncthreads();   // B tile for this step visible to all waves

        // A fragment: VGPR0-3 <- K = half*16..+7 ; VGPR4-7 <- K = half*16+8..+15
        v8bf alo = *(const v8bf*)(arow + kk + half * 16);
        v8bf ahi = *(const v8bf*)(arow + kk + half * 16 + 8);
        v16bf a = __builtin_shufflevector(alo, ahi,
                    0,1,2,3,4,5,6,7,8,9,10,11,12,13,14,15);

        // B fragments from LDS: column n = jj*16+l16, K = half*16..+15
        const __bf16* bb = &bsmem[cur][0];
        v16bf b0 = *(const v16bf*)(bb + ( 0 + l16) * 32 + half * 16);
        v16bf b1 = *(const v16bf*)(bb + (16 + l16) * 32 + half * 16);
        v16bf b2 = *(const v16bf*)(bb + (32 + l16) * 32 + half * 16);
        v16bf b3 = *(const v16bf*)(bb + (48 + l16) * 32 + half * 16);

        acc[0] = __builtin_amdgcn_wmma_f32_16x16x32_bf16(false, a, false, b0,
                                                         (short)0, acc[0], false, false);
        acc[1] = __builtin_amdgcn_wmma_f32_16x16x32_bf16(false, a, false, b1,
                                                         (short)0, acc[1], false, false);
        acc[2] = __builtin_amdgcn_wmma_f32_16x16x32_bf16(false, a, false, b2,
                                                         (short)0, acc[2], false, false);
        acc[3] = __builtin_amdgcn_wmma_f32_16x16x32_bf16(false, a, false, b3,
                                                         (short)0, acc[3], false, false);

        __syncthreads();   // all waves done reading before buffer reuse
    }

    // Epilogue. C layout: lane -> col n = l16, VGPR r -> row = r + 8*half.
    #pragma unroll
    for (int jj = 0; jj < 4; ++jj) {
        const int col = jBase + jj * 16 + l16;
        const float bs = bias[col];
        float csum = 0.0f;
        #pragma unroll
        for (int r = 0; r < 8; ++r) {
            const int row = mBase + r + half * 8;
            float v = acc[jj][r] + bs;
            if (resid) v += resid[(size_t)row * D_FEAT + col];
            if (out)   out[(size_t)row * D_FEAT + col] = v;
            csum += v;
        }
        if (colsum) unsafeAtomicAdd(&colsum[col], csum);
    }
}

// ---------------------------------------------------------------------------
// 6. head: pred[c] = tanh( dot(colsum/N, head_w[c]) + head_b[c] ), c=0,1
// ---------------------------------------------------------------------------
__global__ void head_kernel(const float* __restrict__ colsum,
                            const float* __restrict__ hw,
                            const float* __restrict__ hb,
                            float* __restrict__ out) {
    __shared__ float s0[128];
    __shared__ float s1[128];
    int t = threadIdx.x;
    float a0 = 0.0f, a1 = 0.0f;
    const float invN = 1.0f / (float)N_NODES;
    for (int c = t; c < D_FEAT; c += 128) {
        float m = colsum[c] * invN;
        a0 += m * hw[c];
        a1 += m * hw[D_FEAT + c];
    }
    s0[t] = a0; s1[t] = a1;
    __syncthreads();
    for (int s = 64; s > 0; s >>= 1) {
        if (t < s) { s0[t] += s0[t + s]; s1[t] += s1[t + s]; }
        __syncthreads();
    }
    if (t == 0) {
        out[0] = tanhf(s0[0] + hb[0]);
        out[1] = tanhf(s1[0] + hb[1]);
    }
}

// ---------------------------------------------------------------------------
extern "C" void kernel_launch(void* const* d_in, const int* in_sizes, int n_in,
                              void* d_out, int out_size, void* d_ws, size_t ws_size,
                              hipStream_t stream) {
    const float* lm     = (const float*)d_in[0];   // [N,1024]
    const float* nf     = (const float*)d_in[1];   // [N,256]
    const float* ef     = (const float*)d_in[2];   // [E]
    const int*   src    = (const int*)  d_in[3];   // [E]
    const int*   dst    = (const int*)  d_in[4];   // [E]
    const float* gin_w  = (const float*)d_in[5];   // [D,D]
    const float* gin_b  = (const float*)d_in[6];   // [D]
    const float* gin1_w = (const float*)d_in[7];   // [D,D]
    const float* gin1_b = (const float*)d_in[8];   // [D]
    const float* head_w = (const float*)d_in[9];   // [2,D]
    const float* head_b = (const float*)d_in[10];  // [2]
    float* out = (float*)d_out;

    // workspace carve (all 256B aligned)
    char* ws = (char*)d_ws;
    size_t off = 0;
    const size_t ND4 = (size_t)N_NODES * D_FEAT * sizeof(float);   // 83,886,080
    const size_t ND2 = (size_t)N_NODES * D_FEAT * sizeof(__bf16);  // 41,943,040
    const size_t DD2 = (size_t)D_FEAT * D_FEAT * sizeof(__bf16);   //  3,276,800
    float*  feat0  = (float*)(ws + off); off += ND4;
    float*  agg    = (float*)(ws + off); off += ND4;
    float*  h1     = (float*)(ws + off); off += ND4;
    __bf16* xbf    = (__bf16*)(ws + off); off += ND2;
    __bf16* wbf    = (__bf16*)(ws + off); off += DD2;
    float*  colsum = (float*)(ws + off); off += D_FEAT * sizeof(float);

    const int ND    = N_NODES * D_FEAT;        // 20,971,520
    const int ND_v4 = ND / 4;                  //  5,242,880
    const int DD_v4 = (D_FEAT * D_FEAT) / 4;   //    409,600

    dim3 gemmGrid(N_NODES / 128, D_FEAT / 64); // (128, 20)

    // 1. concat
    concat_kernel<<<(N_NODES * 320) / 256, 256, 0, stream>>>(lm, nf, feat0);
    // zero colsum (for GEMM2 epilogue accumulation)
    zero_kernel<<<(D_FEAT + 255) / 256, 256, 0, stream>>>(colsum, D_FEAT);

    // ---- conv 1 ----
    zero_kernel<<<ND / 256, 256, 0, stream>>>(agg, ND);
    scatter_kernel<<<N_EDGES, 320, 0, stream>>>(feat0, src, dst, ef, agg);
    addcvt_kernel<<<ND_v4 / 256, 256, 0, stream>>>(feat0, agg, xbf);
    cvt_kernel<<<DD_v4 / 256, 256, 0, stream>>>(gin_w, wbf);
    gemm_bf16_kernel<<<gemmGrid, 256, 0, stream>>>(xbf, wbf, gin_b,
                                                   h1, nullptr, nullptr);

    // ---- conv 2 (fused residual + column-sum epilogue, no C store) ----
    zero_kernel<<<ND / 256, 256, 0, stream>>>(agg, ND);
    scatter_kernel<<<N_EDGES, 320, 0, stream>>>(h1, src, dst, ef, agg);
    addcvt_kernel<<<ND_v4 / 256, 256, 0, stream>>>(h1, agg, xbf);
    cvt_kernel<<<DD_v4 / 256, 256, 0, stream>>>(gin1_w, wbf);
    gemm_bf16_kernel<<<gemmGrid, 256, 0, stream>>>(xbf, wbf, gin1_b,
                                                   nullptr, feat0, colsum);

    // ---- head ----
    head_kernel<<<1, 128, 0, stream>>>(colsum, head_w, head_b, out);
}
